// STULayer_6262062318086
// MI455X (gfx1250) — matrix operand order repaired and checked by reading, not compiled, not run on hardware
//
#include <hip/hip_runtime.h>
#include <hip/hip_bf16.h>

typedef __attribute__((ext_vector_type(16))) __bf16 v16bf;
typedef __attribute__((ext_vector_type(8)))  __bf16 v8bf;
typedef __attribute__((ext_vector_type(8)))  float  v8f;

#define HH 4
#define AA 64
#define VV 64
#define DD 512      // model dim
#define UO 1024     // uvqk width = (2V+2A)*H
#define OI 1024     // concat width = 2*V*H + D
#define HV 256      // H*V

static __device__ __forceinline__ int imin(int a, int b) { return a < b ? a : b; }

static __device__ __forceinline__ float silu_f(float x) {
    // x * sigmoid(x) with HW v_rcp_f32 (avoids IEEE div chain)
    return x * __builtin_amdgcn_rcpf(1.0f + __expf(-x));
}

// A-fragment (16x32 bf16): per lane 16 contiguous k (k = 16h + s), two 16B chunks
static __device__ __forceinline__ v16bf load_a16(const __bf16* rp, int kha) {
    v16bf a;
    ((v8bf*)&a)[0] = *(const v8bf*)(rp + kha);
    ((v8bf*)&a)[1] = *(const v8bf*)(rp + 8 + kha);
    return a;
}

// ---------------------------------------------------------------------------
// Kernel 1: input LayerNorm -> nx (bf16), and raw x -> bf16 copy
// ---------------------------------------------------------------------------
__global__ __launch_bounds__(256)
void input_ln_kernel(const float* __restrict__ x,
                     const float* __restrict__ w,
                     const float* __restrict__ b,
                     __bf16* __restrict__ nx,
                     __bf16* __restrict__ xbf) {
    __shared__ float ssum[256];
    __shared__ float ssq[256];
    const int row = blockIdx.x;
    const int tid = threadIdx.x;
    const float* xr = x + (size_t)row * DD;
    float v0 = xr[tid];
    float v1 = xr[tid + 256];
    xbf[(size_t)row * DD + tid]       = (__bf16)v0;
    xbf[(size_t)row * DD + tid + 256] = (__bf16)v1;
    ssum[tid] = v0 + v1; ssq[tid] = v0 * v0 + v1 * v1;
    __syncthreads();
    for (int st = 128; st > 0; st >>= 1) {
        if (tid < st) { ssum[tid] += ssum[tid + st]; ssq[tid] += ssq[tid + st]; }
        __syncthreads();
    }
    const float mu  = ssum[0] * (1.0f / DD);
    const float var = ssq[0] * (1.0f / DD) - mu * mu;
    const float rn  = rsqrtf(var + 1e-6f);
    nx[(size_t)row * DD + tid]       = (__bf16)((v0 - mu) * rn * w[tid] + b[tid]);
    nx[(size_t)row * DD + tid + 256] = (__bf16)((v1 - mu) * rn * w[tid + 256] + b[tid + 256]);
}

// ---------------------------------------------------------------------------
// Kernel 2: pack W (K x N, row-major f32) into B-fragment-major bf16:
// dst[((k/32)*(N/16) + n/16)*512 + lane*16 + i], lane = (n%16) + ((k%32)/16)*16,
// i = k%16  =>  each lane's 16 values contiguous (two b128 per fragment).
// ---------------------------------------------------------------------------
__global__ __launch_bounds__(256)
void pack_b_kernel(const float* __restrict__ src, __bf16* __restrict__ dst,
                   int K, int N) {
    const int f = blockIdx.x * 256 + threadIdx.x;
    if (f >= K * N) return;
    const int i    = f & 15;
    const int lane = (f >> 4) & 31;
    const int tile = f >> 9;
    const int ntiles = N >> 4;
    const int nt = tile % ntiles;
    const int kt = tile / ntiles;
    const int k = (kt << 5) + ((lane >> 4) << 4) + i;
    const int n = (nt << 4) + (lane & 15);
    dst[f] = (__bf16)src[(size_t)k * N + n];
}

// ---------------------------------------------------------------------------
// Kernel 3: uvqk = silu(nx @ Wuvqk + bias); 1 wave per 32x64 tile.
// Also scatters the v slice (cols [256,512)) into token-major vT for attention.
// ---------------------------------------------------------------------------
template<bool FULL>
static __device__ __forceinline__ void store_uvqk_tile(
    const v8f (&acc)[2][4], const float* __restrict__ bias,
    __bf16* __restrict__ Cout, __bf16* __restrict__ vT,
    int rowbase, int nt, int cl, int ch, int M, int Lpad) {
#pragma unroll
    for (int rb = 0; rb < 2; ++rb)
#pragma unroll
        for (int j = 0; j < 4; ++j) {
            const int col = nt * 64 + j * 16 + cl;
            const float bs = bias[col];
#pragma unroll
            for (int r = 0; r < 8; ++r) {
                const int row = rowbase + rb * 16 + r + 8 * ch;
                if (FULL || row < M) {
                    const __bf16 val = (__bf16)silu_f(acc[rb][j][r] + bs);
                    Cout[(size_t)row * UO + col] = val;
                    if (col >= HV && col < 2 * HV)  // v slice -> transposed copy
                        vT[(size_t)(col - HV) * Lpad + row] = val;
                }
            }
        }
}

__global__ __launch_bounds__(32)
void uvqk_gemm_kernel(const __bf16* __restrict__ Abf,
                      const __bf16* __restrict__ Bpk,
                      const float*  __restrict__ bias,
                      __bf16* __restrict__ Cout,
                      __bf16* __restrict__ vT,
                      int M, int Lpad) {
    const int mt = blockIdx.x, nt = blockIdx.y;
    const int lane = threadIdx.x & 31;
    const int cl = lane & 15;
    const int ch = lane >> 4;
    const int kha = ch * 16;
    v8f acc[2][4] = {};
    const int rowbase = mt * 32;
    const int r0 = imin(rowbase + cl, M - 1);
    const int r1 = imin(rowbase + 16 + cl, M - 1);
    const int NT16 = UO / 16;
    for (int kt = 0; kt < DD; kt += 32) {
        const v16bf a0 = load_a16(Abf + (size_t)r0 * DD + kt, kha);
        const v16bf a1 = load_a16(Abf + (size_t)r1 * DD + kt, kha);
        const __bf16* bp = Bpk + ((size_t)(kt >> 5) * NT16 + nt * 4) * 512 + lane * 16;
#pragma unroll
        for (int j = 0; j < 4; ++j) {
            const v16bf bf = *(const v16bf*)(bp + j * 512);
            acc[0][j] = __builtin_amdgcn_wmma_f32_16x16x32_bf16(
                false, a0, false, bf, (short)0, acc[0][j], false, false);
            acc[1][j] = __builtin_amdgcn_wmma_f32_16x16x32_bf16(
                false, a1, false, bf, (short)0, acc[1][j], false, false);
        }
    }
    if (rowbase + 31 < M)
        store_uvqk_tile<true>(acc, bias, Cout, vT, rowbase, nt, cl, ch, M, Lpad);
    else
        store_uvqk_tile<false>(acc, bias, Cout, vT, rowbase, nt, cl, ch, M, Lpad);
}

// ---------------------------------------------------------------------------
// Kernel 4: masked SiLU attention, 1 wave per (b,h,16-row tile)
// uvqk column slices: u[0,256) v[256,512) q[512,768) k[768,1024)
// ---------------------------------------------------------------------------
__global__ __launch_bounds__(32)
void attn_kernel(const __bf16* __restrict__ uvqk,
                 const __bf16* __restrict__ vT,
                 const int* __restrict__ lengths,
                 const int* __restrict__ offsets,
                 const int* __restrict__ ntgt,
                 const int* __restrict__ pN,
                 float* __restrict__ attn,
                 int L, int Lpad) {
    __shared__ __bf16 swb[16 * 32];
    const int b = blockIdx.z, h = blockIdx.y, nt = blockIdx.x;
    const int len = lengths[b];
    const int n0 = nt * 16;
    if (n0 >= len) return;
    const int off = offsets[b];
    const int maxid = len - ntgt[b];
    const float invN = 1.0f / (float)pN[0];
    const float alpha = 0.125f;  // A^{-1/2}, A=64

    const int lane = threadIdx.x & 31;
    const int cl = lane & 15;
    const int ch = lane >> 4;
    const int kha = ch * 16;
    const int khb = ch * 16;

    // Q A-fragments (16x64 split into two 16x32 frags)
    const int qtok = imin(off + n0 + cl, L - 1);
    const __bf16* qrp = uvqk + (size_t)qtok * UO + 2 * HV + h * AA;
    const v16bf qa0 = load_a16(qrp, kha);
    const v16bf qa1 = load_a16(qrp + 32, kha);

    v8f acc[4] = {};
    const int mmax = imin(n0 + 15, len - 1);
    const int nmb = (mmax >> 5) + 1;

    for (int mb = 0; mb < nmb; ++mb) {
        const int m0 = mb * 32;
        // ---- scores for two 16-wide m subtiles ----
#pragma unroll
        for (int ms = 0; ms < 2; ++ms) {
            const int mc = m0 + ms * 16;
            const int ktok = imin(off + mc + cl, L - 1);
            const __bf16* krp = uvqk + (size_t)ktok * UO + 2 * HV + HH * AA + h * AA;
            const v16bf kb0 = *(const v16bf*)(krp + khb);
            const v16bf kb1 = *(const v16bf*)(krp + 32 + khb);
            v8f s = {};
            s = __builtin_amdgcn_wmma_f32_16x16x32_bf16(false, qa0, false, kb0, (short)0, s, false, false);
            s = __builtin_amdgcn_wmma_f32_16x16x32_bf16(false, qa1, false, kb1, (short)0, s, false, false);

            const int m_ = mc + cl;
            const int idm = imin(m_, maxid);
            const bool vm = m_ < len;
#pragma unroll
            for (int r = 0; r < 8; ++r) {
                const int row = r + 8 * ch;
                const int n_ = n0 + row;
                const int idn = imin(n_, maxid);
                const bool keep = ((idn > idm) || (n_ == m_)) && vm && (n_ < len);
                const float xv = s[r] * alpha;
                swb[row * 32 + ms * 16 + cl] = (__bf16)(keep ? silu_f(xv) * invN : 0.0f);
            }
        }
        __syncthreads();
        // ---- scores as bf16 A-fragment (16x32, K = m-local): two ds_load_b128 ----
        v16bf wa;
        ((v8bf*)&wa)[0] = *(const v8bf*)(swb + cl * 32 + kha);
        ((v8bf*)&wa)[1] = *(const v8bf*)(swb + cl * 32 + 8 + kha);
        __syncthreads();
        // ---- P @ V from token-major vT: contiguous 32B per lane ----
#pragma unroll
        for (int vj = 0; vj < 4; ++vj) {
            const __bf16* vp = vT + (size_t)(h * VV + vj * 16 + cl) * Lpad + off + m0 + khb;
            const v16bf vb = *(const v16bf*)vp;
            acc[vj] = __builtin_amdgcn_wmma_f32_16x16x32_bf16(
                false, wa, false, vb, (short)0, acc[vj], false, false);
        }
    }
    // ---- store attn_p[tok, h*64 + vcol] (f32) ----
#pragma unroll
    for (int r = 0; r < 8; ++r) {
        const int n_ = n0 + r + 8 * ch;
        if (n_ < len) {
            float* op = attn + (size_t)(off + n_) * HV + h * VV;
            op[0 * 16 + cl] = acc[0][r];
            op[1 * 16 + cl] = acc[1][r];
            op[2 * 16 + cl] = acc[2][r];
            op[3 * 16 + cl] = acc[3][r];
        }
    }
}

// ---------------------------------------------------------------------------
// Kernel 5: output LayerNorm (over 256) + assemble Y = [u | x | y_norm] (bf16)
// ---------------------------------------------------------------------------
__global__ __launch_bounds__(256)
void outnorm_assemble_kernel(const float* __restrict__ attn,
                             const float* __restrict__ w,
                             const float* __restrict__ b,
                             const __bf16* __restrict__ uvqk,
                             const __bf16* __restrict__ xbf,
                             __bf16* __restrict__ Y) {
    __shared__ float ssum[256];
    __shared__ float ssq[256];
    const int tok = blockIdx.x;
    const int tid = threadIdx.x;
    const float a = attn[(size_t)tok * HV + tid];
    ssum[tid] = a; ssq[tid] = a * a;
    __syncthreads();
    for (int st = 128; st > 0; st >>= 1) {
        if (tid < st) { ssum[tid] += ssum[tid + st]; ssq[tid] += ssq[tid + st]; }
        __syncthreads();
    }
    const float mu  = ssum[0] * (1.0f / HV);
    const float var = ssq[0] * (1.0f / HV) - mu * mu;
    const float y = (a - mu) * rsqrtf(var + 1e-6f) * w[tid] + b[tid];
    __bf16* yr = Y + (size_t)tok * OI;
    yr[tid]       = uvqk[(size_t)tok * UO + tid];       // u
    yr[256 + tid] = xbf[(size_t)tok * DD + tid];        // x lo
    yr[512 + tid] = xbf[(size_t)tok * DD + 256 + tid];  // x hi
    yr[768 + tid] = (__bf16)y;                          // y_norm
}

// ---------------------------------------------------------------------------
// Kernel 6: out = x + Y @ Wout; 1 wave per 32x64 tile, f32 out
// ---------------------------------------------------------------------------
template<bool FULL>
static __device__ __forceinline__ void store_out_tile(
    const v8f (&acc)[2][4], const float* __restrict__ x,
    float* __restrict__ out, int rowbase, int nt, int cl, int ch, int M) {
#pragma unroll
    for (int rb = 0; rb < 2; ++rb)
#pragma unroll
        for (int j = 0; j < 4; ++j) {
            const int col = nt * 64 + j * 16 + cl;
#pragma unroll
            for (int r = 0; r < 8; ++r) {
                const int row = rowbase + rb * 16 + r + 8 * ch;
                if (FULL || row < M)
                    out[(size_t)row * DD + col] = x[(size_t)row * DD + col] + acc[rb][j][r];
            }
        }
}

__global__ __launch_bounds__(32)
void final_gemm_kernel(const __bf16* __restrict__ Ybf,
                       const __bf16* __restrict__ Bpk,
                       const float*  __restrict__ x,
                       float* __restrict__ out,
                       int M) {
    const int mt = blockIdx.x, nt = blockIdx.y;
    const int lane = threadIdx.x & 31;
    const int cl = lane & 15;
    const int ch = lane >> 4;
    const int kha = ch * 16;
    v8f acc[2][4] = {};
    const int rowbase = mt * 32;
    const int r0 = imin(rowbase + cl, M - 1);
    const int r1 = imin(rowbase + 16 + cl, M - 1);
    const int NT16 = DD / 16;
    for (int kt = 0; kt < OI; kt += 32) {
        const v16bf a0 = load_a16(Ybf + (size_t)r0 * OI + kt, kha);
        const v16bf a1 = load_a16(Ybf + (size_t)r1 * OI + kt, kha);
        const __bf16* bp = Bpk + ((size_t)(kt >> 5) * NT16 + nt * 4) * 512 + lane * 16;
#pragma unroll
        for (int j = 0; j < 4; ++j) {
            const v16bf bf = *(const v16bf*)(bp + j * 512);
            acc[0][j] = __builtin_amdgcn_wmma_f32_16x16x32_bf16(
                false, a0, false, bf, (short)0, acc[0][j], false, false);
            acc[1][j] = __builtin_amdgcn_wmma_f32_16x16x32_bf16(
                false, a1, false, bf, (short)0, acc[1][j], false, false);
        }
    }
    if (rowbase + 31 < M)
        store_out_tile<true>(acc, x, out, rowbase, nt, cl, ch, M);
    else
        store_out_tile<false>(acc, x, out, rowbase, nt, cl, ch, M);
}

// ---------------------------------------------------------------------------
extern "C" void kernel_launch(void* const* d_in, const int* in_sizes, int n_in,
                              void* d_out, int out_size, void* d_ws, size_t ws_size,
                              hipStream_t stream) {
    const float* x      = (const float*)d_in[0];
    const int*   xlen   = (const int*)d_in[1];
    const int*   xoff   = (const int*)d_in[2];
    const int*   pN     = (const int*)d_in[3];
    const int*   ntg    = (const int*)d_in[4];
    const float* wuvqk  = (const float*)d_in[5];
    const float* buvqk  = (const float*)d_in[6];
    const float* inw    = (const float*)d_in[7];
    const float* inb    = (const float*)d_in[8];
    const float* onw    = (const float*)d_in[9];
    const float* onb    = (const float*)d_in[10];
    const float* wout   = (const float*)d_in[11];

    const int L = in_sizes[0] / DD;
    const int B = in_sizes[1];
    const int Lpad = ((L + 31) & ~31) + 64;  // token-major vT row pitch (OOB-safe tail)

    char* wp = (char*)d_ws;
    auto take = [&](size_t bytes) -> void* {
        void* p = (void*)wp;
        wp += (bytes + 255) & ~(size_t)255;
        return p;
    };
    __bf16* nx   = (__bf16*)take((size_t)L * DD * 2);
    __bf16* xbf  = (__bf16*)take((size_t)L * DD * 2);
    __bf16* wub  = (__bf16*)take((size_t)DD * UO * 2);
    __bf16* wob  = (__bf16*)take((size_t)OI * DD * 2);
    __bf16* uvb  = (__bf16*)take((size_t)L * UO * 2);
    __bf16* vT   = (__bf16*)take((size_t)HV * Lpad * 2);
    float*  attn = (float*)take((size_t)L * HV * 4);
    __bf16* Ybf  = (__bf16*)take((size_t)L * OI * 2);

    input_ln_kernel<<<L, 256, 0, stream>>>(x, inw, inb, nx, xbf);
    pack_b_kernel<<<(DD * UO + 255) / 256, 256, 0, stream>>>(wuvqk, wub, DD, UO);
    pack_b_kernel<<<(OI * DD + 255) / 256, 256, 0, stream>>>(wout, wob, OI, DD);

    const int Mt32 = (L + 31) / 32;
    dim3 g1(Mt32, UO / 64);
    uvqk_gemm_kernel<<<g1, 32, 0, stream>>>(nx, wub, buvqk, uvb, vT, L, Lpad);

    const int Mt16 = (L + 15) / 16;
    dim3 g2(Mt16, HH, B);
    attn_kernel<<<g2, 32, 0, stream>>>(uvb, vT, xlen, xoff, ntg, pN, attn, L, Lpad);

    outnorm_assemble_kernel<<<L, 256, 0, stream>>>(attn, onw, onb, uvb, xbf, Ybf);

    dim3 g3(Mt32, DD / 64);
    final_gemm_kernel<<<g3, 32, 0, stream>>>(Ybf, wob, x, (float*)d_out, L);
}
